// NoisyGRUSeq2SeqWithFeatures_19387482374760
// MI455X (gfx1250) — matrix operand order repaired and checked by reading, not compiled, not run on hardware
//
#include <hip/hip_runtime.h>
#include <hip/hip_bf16.h>
#include <stdint.h>

// ---------------------------------------------------------------------------
// CDNA5 (gfx1250) persistent GRU stack, v_wmma_f32_16x16x32_bf16.
// B=64, T=128, layers 512/1024/2048, dense head 3584->512 with tanh.
// Each wave owns one 16-column n-tile across all 4 M-tiles (B loaded once,
// 4 independent WMMA accumulator chains). A activations are staged into LDS
// with DOUBLE-BUFFERED global_load_async_to_lds_b128: chunk c+1 is fetched
// under the WMMA stream of chunk c (ASYNCcnt pipelining).
// ---------------------------------------------------------------------------

typedef __attribute__((ext_vector_type(16))) __bf16 v16bf;
typedef __attribute__((ext_vector_type(8)))  float  v8f;

#define BATCH 64
#define TLEN  128
#define NBLK  32        // persistent grid (device-wide barrier participants)
#define CH    8         // K-chunk in 32-wide k-blocks -> 8*4*1024B = 32KB/buffer
#define CHBYTES (CH * 4 * 1024)
#define CHELEMS (CH * 4 * 512)

// ---- CDNA5 WMMA lane layouts (ISA 7.12.2) ---------------------------------
// A (16x32 bf16 MxK): lane l, elem j: M=l%16, K=(j&7)+16*(j>=8)+8*(l>=16)
__device__ __forceinline__ int aswz(int m, int k, int Kbuf) {
  int kk   = k & 31;
  int hi   = (kk >> 3) & 1;
  int j    = (kk & 7) + ((kk & 16) ? 8 : 0);
  int lane = (m & 15) | (hi << 4);
  int blk  = (m >> 4) * (Kbuf >> 5) + (k >> 5);
  return (blk << 9) + (lane << 4) + j;          // bf16 elements
}
// B (32x16 bf16 KxN): lane l, elem j: N=l%16, K=(j&7)+16*(j>=8)+8*(l>=16)
__device__ __forceinline__ int bswz(int k, int n, int Kbuf) {
  int kk   = k & 31;
  int lane = (n & 15) | (((kk >> 3) & 1) << 4);
  int j    = (kk & 7) + ((kk & 16) ? 8 : 0);
  int blk  = (n >> 4) * (Kbuf >> 5) + (k >> 5); // N-tile major: K streams contiguously
  return (blk << 9) + (lane << 4) + j;
}

struct APart { const __bf16* p; int kb; };      // kb = #32-wide K blocks

// issue async copies of one K-chunk (clen k-blocks x 4 m-tiles x 1024B) to LDS
__device__ __forceinline__ void stage_chunk(const APart* parts, int c, int clen,
                                            unsigned ldsbase, int tid) {
  for (int it = 0; it < clen; ++it) {
    int i = it * 256 + tid;                     // blockDim.x == 256
    int c16 = i & 63, mt = (i >> 6) & 3, kbl = i >> 8;
    int kb = c + kbl;
    const APart* pp = parts; int kloc = kb;
    while (kloc >= pp->kb) { kloc -= pp->kb; ++pp; }
    const __bf16* src = pp->p + (((size_t)mt * pp->kb + kloc) << 9) + (c16 << 3);
    unsigned ldsoff = ldsbase + (unsigned)((((kbl << 2) + mt) << 10) + (c16 << 4));
    asm volatile("global_load_async_to_lds_b128 %0, %1, off"
                 :: "v"(ldsoff), "v"(src) : "memory");
  }
}

// ---- staged GEMM: acc[mt] (16x16 f32) over all 4 M-tiles of one n-tile ----
__device__ __forceinline__ void staged_mm(__bf16* As, const APart* parts, int np,
                                          const __bf16* W, int ktot, int Ntiles,
                                          const float* bias, int wave, int lane,
                                          int tid, v8f acc[4]) {
  const int  nt     = wave;
  const bool active = nt < Ntiles;
  const float bv    = active ? bias[(nt << 4) + (lane & 15)] : 0.f;
#pragma unroll
  for (int mt = 0; mt < 4; ++mt)
#pragma unroll
    for (int i = 0; i < 8; ++i) acc[mt][i] = bv;   // bias folded into C

  const v16bf* Bp = (const v16bf*)W + (size_t)nt * ktot * 32 + lane;
  const int nch = (ktot + CH - 1) / CH;

  // prologue: prefetch chunk 0 into buffer 0 (prior block-wide sync done by gsync)
  stage_chunk(parts, 0, (ktot < CH ? ktot : CH), 0u, tid);

  unsigned buf = 0;
  for (int ci = 0; ci < nch; ++ci) {
    const int c = ci * CH;
    const int clen = (ktot - c < CH) ? (ktot - c) : CH;
    asm volatile("s_wait_asynccnt 0x0" ::: "memory");  // my async copies done
    __syncthreads();                                   // buffer full; old buffer free
    if (ci + 1 < nch) {                                // fetch next chunk under compute
      const int c2 = c + CH;
      stage_chunk(parts, c2, (ktot - c2 < CH) ? (ktot - c2) : CH,
                  (buf ^ 1u) * (unsigned)CHBYTES, tid);
    }
    if (active) {
      const __bf16* Ab = As + (size_t)buf * CHELEMS;
      for (int i = 0; i < clen; ++i) {
        v16bf b = Bp[(size_t)(c + i) * 32];
        __builtin_prefetch((const void*)(Bp + (size_t)(c + i + 8) * 32), 0, 1);
#pragma unroll
        for (int mt = 0; mt < 4; ++mt) {
          v16bf a = *(const v16bf*)(Ab + ((((i << 2) + mt) << 9) + (lane << 4)));
          acc[mt] = __builtin_amdgcn_wmma_f32_16x16x32_bf16(false, a, false, b,
                                                            (short)0, acc[mt],
                                                            false, false);
        }
      }
    }
    buf ^= 1u;
  }
}

// ---- device-wide barrier (generation counter) -----------------------------
__device__ __forceinline__ void gsync(unsigned* bar, unsigned nb) {
  __threadfence();
  __syncthreads();
  if (threadIdx.x == 0) {
    unsigned gen = __hip_atomic_load(&bar[1], __ATOMIC_ACQUIRE, __HIP_MEMORY_SCOPE_AGENT);
    unsigned arrived = __hip_atomic_fetch_add(&bar[0], 1u, __ATOMIC_ACQ_REL, __HIP_MEMORY_SCOPE_AGENT);
    if (arrived == nb - 1u) {
      __hip_atomic_store(&bar[0], 0u, __ATOMIC_RELAXED, __HIP_MEMORY_SCOPE_AGENT);
      __hip_atomic_fetch_add(&bar[1], 1u, __ATOMIC_ACQ_REL, __HIP_MEMORY_SCOPE_AGENT);
    } else {
      while (__hip_atomic_load(&bar[1], __ATOMIC_ACQUIRE, __HIP_MEMORY_SCOPE_AGENT) == gen)
        __builtin_amdgcn_s_sleep(2);
    }
  }
  __syncthreads();
}

// ---- prep kernels ----------------------------------------------------------
__global__ void zero_mem(unsigned* p, int n) {
  int i = blockIdx.x * blockDim.x + threadIdx.x;
  if (i < n) p[i] = 0u;
}

__global__ void swz_weights(const float* __restrict__ src, __bf16* __restrict__ dst,
                            int K, int N, int transposed) {
  long idx = (long)blockIdx.x * blockDim.x + threadIdx.x;
  long total = (long)K * N;
  if (idx >= total) return;
  int k = (int)(idx / N), n = (int)(idx % N);
  float v = transposed ? src[(long)n * K + k] : src[(long)k * N + n];
  dst[bswz(k, n, K)] = (__bf16)v;
}

__global__ void embed_swz(const int* __restrict__ seqs, const float* __restrict__ emb,
                          __bf16* __restrict__ Xemb) {
  int idx = blockIdx.x * blockDim.x + threadIdx.x;
  if (idx >= TLEN * BATCH * 32) return;
  int t = idx >> 11;
  int r = idx & 2047;
  int b = r >> 5, e = r & 31;
  int tok = seqs[b * TLEN + t];
  float v = emb[tok * 32 + e];
  Xemb[(size_t)t * (BATCH * 32) + aswz(b, e, 32)] = (__bf16)v;
}

// ---- persistent GRU kernel -------------------------------------------------
struct Params {
  const int* lens;
  const float* GB[3]; const float* CB[3]; const float* dense_b;
  const __bf16* Wg[3]; const __bf16* Wc[3]; const __bf16* Wd;
  const __bf16* Xemb;
  __bf16* Xb[3];     // raw layer outputs (A-swizzled bf16); [0] unused
  __bf16* Hb[3];     // masked hidden state, bf16 A-swizzled
  __bf16* RH[3];     // r*h, bf16 A-swizzled
  float*  Hf[3];     // masked hidden state, f32 master (row-major)
  float*  Z;         // z gate scratch, stride 2048
  float*  out;
  unsigned* bar;
  unsigned nblocks;
};

__global__ __launch_bounds__(256) void gru_persistent(Params P) {
  extern __shared__ char smem[];
  __bf16* As = (__bf16*)smem;
  const int tid  = threadIdx.x;
  const int lane = tid & 31;
  const int wave = blockIdx.x * (blockDim.x >> 5) + (tid >> 5);
  const int IN[3]  = {32, 512, 1024};
  const int OUT[3] = {512, 1024, 2048};
  v8f acc[4];

  for (int t = 0; t < TLEN; ++t) {
    for (int L = 0; L < 3; ++L) {
      const int ins = IN[L], outs = OUT[L];
      const int xkb = ins >> 5, hkb = outs >> 5, ktot = xkb + hkb;
      const __bf16* Xp = (L == 0) ? (P.Xemb + (size_t)t * (BATCH * 32)) : P.Xb[L];

      // ---- gates = sigmoid([x,h] @ Kg + bg) -> r*h, z ----
      {
        const int Ntiles = (2 * outs) >> 4;
        APart parts[2] = { {Xp, xkb}, {P.Hb[L], hkb} };
        staged_mm(As, parts, 2, P.Wg[L], ktot, Ntiles, P.GB[L], wave, lane, tid, acc);
        if (wave < Ntiles) {
          int n = (wave << 4) + (lane & 15);
#pragma unroll
          for (int mt = 0; mt < 4; ++mt)
#pragma unroll
            for (int v = 0; v < 8; ++v) {
              float g = 1.f / (1.f + __expf(-acc[mt][v]));
              int m = (mt << 4) + v + ((lane >> 4) << 3);
              if (n < outs) {                               // r gate -> r*h
                float h = P.Hf[L][m * outs + n];
                P.RH[L][aswz(m, n, outs)] = (__bf16)(g * h);
              } else {                                      // z gate
                P.Z[m * 2048 + (n - outs)] = g;
              }
            }
        }
      }
      gsync(P.bar, P.nblocks);

      // ---- cand = tanh([x, r*h] @ Kc + cb); h' = z*h + (1-z)*cand; mask ----
      {
        const int Ntiles = outs >> 4;
        APart parts[2] = { {Xp, xkb}, {P.RH[L], hkb} };
        staged_mm(As, parts, 2, P.Wc[L], ktot, Ntiles, P.CB[L], wave, lane, tid, acc);
        if (wave < Ntiles) {
          int n = (wave << 4) + (lane & 15);
#pragma unroll
          for (int mt = 0; mt < 4; ++mt)
#pragma unroll
            for (int v = 0; v < 8; ++v) {
              float c  = tanhf(acc[mt][v]);
              int m    = (mt << 4) + v + ((lane >> 4) << 3);
              float z  = P.Z[m * 2048 + n];
              float hp = P.Hf[L][m * outs + n];
              float xraw = z * hp + (1.f - z) * c;          // raw output (next layer)
              float hnew = (t < P.lens[m]) ? xraw : hp;     // masked carry
              P.Hf[L][m * outs + n]     = hnew;
              P.Hb[L][aswz(m, n, outs)] = (__bf16)hnew;
              if (L < 2) P.Xb[L + 1][aswz(m, n, outs)] = (__bf16)xraw;
            }
        }
      }
      gsync(P.bar, P.nblocks);
    }
  }

  // ---- dense head: out = tanh(concat(h0,h1,h2) @ dense_w.T + dense_b) ----
  {
    APart parts[3] = { {P.Hb[0], 512 >> 5}, {P.Hb[1], 1024 >> 5}, {P.Hb[2], 2048 >> 5} };
    const int ktot = 3584 >> 5;
    const int Ntiles = 512 >> 4;
    staged_mm(As, parts, 3, P.Wd, ktot, Ntiles, P.dense_b, wave, lane, tid, acc);
    if (wave < Ntiles) {
      int n = (wave << 4) + (lane & 15);
#pragma unroll
      for (int mt = 0; mt < 4; ++mt)
#pragma unroll
        for (int v = 0; v < 8; ++v) {
          int m = (mt << 4) + v + ((lane >> 4) << 3);
          P.out[m * 512 + n] = tanhf(acc[mt][v]);
        }
    }
  }
}

// ---------------------------------------------------------------------------
extern "C" void kernel_launch(void* const* d_in, const int* in_sizes, int n_in,
                              void* d_out, int out_size, void* d_ws, size_t ws_size,
                              hipStream_t stream) {
  const int*   seqs = (const int*)d_in[0];
  const int*   lens = (const int*)d_in[1];
  const float* emb  = (const float*)d_in[2];
  const float* gk[3] = {(const float*)d_in[3],  (const float*)d_in[7],  (const float*)d_in[11]};
  const float* gb[3] = {(const float*)d_in[4],  (const float*)d_in[8],  (const float*)d_in[12]};
  const float* ck[3] = {(const float*)d_in[5],  (const float*)d_in[9],  (const float*)d_in[13]};
  const float* cb[3] = {(const float*)d_in[6],  (const float*)d_in[10], (const float*)d_in[14]};
  const float* dw = (const float*)d_in[15];
  const float* db = (const float*)d_in[16];

  const int IN[3]  = {32, 512, 1024};
  const int OUT[3] = {512, 1024, 2048};

  char* ws = (char*)d_ws;
  size_t off = 0;
  auto carve = [&](size_t bytes) -> char* {
    char* p = ws + off;
    off += (bytes + 1023) & ~(size_t)1023;
    return p;
  };

  unsigned* bar = (unsigned*)carve(1024);
  __bf16* Wg[3]; __bf16* Wc[3];
  for (int L = 0; L < 3; ++L) {
    Wg[L] = (__bf16*)carve((size_t)(IN[L] + OUT[L]) * (2 * OUT[L]) * 2);
    Wc[L] = (__bf16*)carve((size_t)(IN[L] + OUT[L]) * OUT[L] * 2);
  }
  __bf16* Wd   = (__bf16*)carve((size_t)3584 * 512 * 2);
  __bf16* Xemb = (__bf16*)carve((size_t)TLEN * BATCH * 32 * 2);
  __bf16* Xb1  = (__bf16*)carve((size_t)BATCH * 512 * 2);
  __bf16* Xb2  = (__bf16*)carve((size_t)BATCH * 1024 * 2);
  __bf16* Hb   = (__bf16*)carve((size_t)BATCH * 3584 * 2);
  __bf16* RH0  = (__bf16*)carve((size_t)BATCH * 512 * 2);
  __bf16* RH1  = (__bf16*)carve((size_t)BATCH * 1024 * 2);
  __bf16* RH2  = (__bf16*)carve((size_t)BATCH * 2048 * 2);
  float*  Hf   = (float*)carve((size_t)BATCH * 3584 * 4);
  float*  Z    = (float*)carve((size_t)BATCH * 2048 * 4);
  if (off > ws_size) return;

  {
    int n1 = 1024 / 4;
    int n2 = (int)((size_t)BATCH * 3584 * 2 / 4);
    int n3 = BATCH * 3584;
    zero_mem<<<(n1 + 255) / 256, 256, 0, stream>>>(bar, n1);
    zero_mem<<<(n2 + 255) / 256, 256, 0, stream>>>((unsigned*)Hb, n2);
    zero_mem<<<(n3 + 255) / 256, 256, 0, stream>>>((unsigned*)Hf, n3);
  }
  for (int L = 0; L < 3; ++L) {
    long tg = (long)(IN[L] + OUT[L]) * (2 * OUT[L]);
    long tc = (long)(IN[L] + OUT[L]) * OUT[L];
    swz_weights<<<(unsigned)((tg + 255) / 256), 256, 0, stream>>>(gk[L], Wg[L], IN[L] + OUT[L], 2 * OUT[L], 0);
    swz_weights<<<(unsigned)((tc + 255) / 256), 256, 0, stream>>>(ck[L], Wc[L], IN[L] + OUT[L], OUT[L], 0);
  }
  {
    long td = (long)3584 * 512;
    swz_weights<<<(unsigned)((td + 255) / 256), 256, 0, stream>>>(dw, Wd, 3584, 512, 1);
  }
  {
    int n = TLEN * BATCH * 32;
    embed_swz<<<(n + 255) / 256, 256, 0, stream>>>(seqs, emb, Xemb);
  }

  Params P;
  P.lens = lens;
  for (int L = 0; L < 3; ++L) { P.GB[L] = gb[L]; P.CB[L] = cb[L]; P.Wg[L] = Wg[L]; P.Wc[L] = Wc[L]; }
  P.dense_b = db; P.Wd = Wd; P.Xemb = Xemb;
  P.Xb[0] = nullptr; P.Xb[1] = Xb1; P.Xb[2] = Xb2;
  P.Hb[0] = Hb;                 P.Hb[1] = Hb + (size_t)BATCH * 512;  P.Hb[2] = Hb + (size_t)BATCH * 1536;
  P.RH[0] = RH0; P.RH[1] = RH1; P.RH[2] = RH2;
  P.Hf[0] = Hf;                 P.Hf[1] = Hf + (size_t)BATCH * 512;  P.Hf[2] = Hf + (size_t)BATCH * 1536;
  P.Z = Z; P.out = (float*)d_out; P.bar = bar; P.nblocks = NBLK;

  gru_persistent<<<NBLK, 256, 65536, stream>>>(P);
}
